// BertAttention_55327768707700
// MI455X (gfx1250) — matrix operand order repaired and checked by reading, not compiled, not run on hardware
//
#include <hip/hip_runtime.h>
#include <hip/hip_bf16.h>

// ---------------------------------------------------------------------------
// BERT attention block on gfx1250 (wave32, WMMA bf16, async-to-LDS staging).
//   B=4, S=2048, H=1024, NH=16, DH=64.
//   1) fp32 -> bf16 converts (WMMA feed precision, fp32 accumulate)
//   2) QKV GEMM: v_wmma_f32_16x16x32_bf16, 128x128 block tile, double-buffered
//      LDS staged via global_load_async_to_lds_b128 (ASYNCcnt DMA path).
//      Q,K stored [b,h,s,d]; V stored transposed [b,h,d,s].
//   3) flash attention: 1 wave / 16 query rows / head, online softmax,
//      scores never hit HBM (avoids the 1 GB probs tensor).
//   4) output projection GEMM (+bias+residual) -> fp32 scratch
//   5) LayerNorm -> d_out
// ---------------------------------------------------------------------------

typedef __bf16 bf16;
typedef __attribute__((ext_vector_type(16))) __bf16 v16bf;
typedef __attribute__((ext_vector_type(8)))  __bf16 v8bf;
typedef __attribute__((ext_vector_type(8)))  float  v8f;

#define BB 4
#define SS 2048
#define HH 1024
#define NHH 16
#define DHH 64
#define MM (BB * SS)          // 8192 rows

static __device__ __forceinline__ v8f wmma_bf16(v16bf a, v16bf b, v8f c) {
    return __builtin_amdgcn_wmma_f32_16x16x32_bf16(
        /*neg_a=*/false, a, /*neg_b=*/false, b,
        /*c_mod=*/(short)0, c, /*reuse_a=*/false, /*reuse_b=*/false);
}

// Async DMA: 16B per lane, global -> LDS, tracked by ASYNCcnt (no VGPR data).
// lds = wave-relative LDS byte address (low 32 bits of the flat address).
static __device__ __forceinline__ void async_b128(unsigned int lds,
                                                  const bf16* g) {
    asm volatile("global_load_async_to_lds_b128 %0, %1, off"
                 :: "v"(lds), "v"(g) : "memory");
}
static __device__ __forceinline__ unsigned int lds_addr(const void* p) {
    return (unsigned int)(unsigned long long)p;  // flat[31:0] == LDS offset
}

// A-matrix 16x32 bf16 fragment from a row-major [rows][32] tile.
// Lane l: M = l&15, half = l>>4; element e -> K = 16*half + e
// (two adjacent 16B chunks).
static __device__ __forceinline__ v16bf load_a_frag32(const bf16* base, int mo,
                                                      int lane, int stride) {
    int half = lane >> 4, row = lane & 15;
    const bf16* p = base + (size_t)(mo + row) * stride;
    v8bf lo = *(const v8bf*)(p + 16 * half);
    v8bf hi = *(const v8bf*)(p + 16 * half + 8);
    return __builtin_shufflevector(lo, hi, 0, 1, 2, 3, 4, 5, 6, 7,
                                   8, 9, 10, 11, 12, 13, 14, 15);
}

// B-matrix 32x16 bf16 fragment from B^T rows (N-major, K contiguous):
// lane l: N = l&15, K = 16*half + e  (one contiguous 32B chunk).
static __device__ __forceinline__ v16bf load_b_frag32(const bf16* base, int no,
                                                      int lane, int stride) {
    int half = lane >> 4, col = lane & 15;
    const bf16* p = base + (size_t)(no + col) * stride + 16 * half;
    return *(const v16bf*)p;
}

// ---------------------------------------------------------------------------
__global__ void __launch_bounds__(256) cvt_f32_bf16(const float* __restrict__ s,
                                                    bf16* __restrict__ d, int n) {
    int i = blockIdx.x * 256 + threadIdx.x;
    if (i < n) d[i] = (bf16)s[i];
}

// ---------------------------------------------------------------------------
// Shared GEMM core: C[128x128] = X[128xK] . W^T[128xK], K=1024, bf16 WMMA.
// 256 threads = 8 waves (4 M x 2 N), wave tile 32x64 (2x4 frags, 8 WMMA/step).
// Double-buffered LDS filled by global_load_async_to_lds_b128.
// Accumulators handed back to the caller's epilogue via acc[2][4].
// ---------------------------------------------------------------------------
#define GEMM_CORE(X, W, m0, n0, acc)                                          \
    __shared__ alignas(32) bf16 sA[2][128 * 32];                              \
    __shared__ alignas(32) bf16 sB[2][128 * 32];                              \
    int t = threadIdx.x, lane = t & 31, wid = t >> 5;                         \
    int wm = wid >> 1, wn = wid & 1;                                          \
    int r0 = t >> 2, c0 = (t & 3) * 8;                                        \
    unsigned int lA[2], lB[2];                                                \
    lA[0] = lds_addr(&sA[0][r0 * 32 + c0]);                                   \
    lA[1] = lds_addr(&sA[1][r0 * 32 + c0]);                                   \
    lB[0] = lds_addr(&sB[0][r0 * 32 + c0]);                                   \
    lB[1] = lds_addr(&sB[1][r0 * 32 + c0]);                                   \
    v8f acc[2][4] = {};                                                       \
    {                                                                         \
        const bf16* ga = X + (size_t)(m0 + r0) * HH + c0;                     \
        const bf16* gb = W + (size_t)(n0 + r0) * HH + c0;                     \
        async_b128(lA[0], ga);                                                \
        async_b128(lA[0] + 64 * 32 * 2, ga + (size_t)64 * HH);                \
        async_b128(lB[0], gb);                                                \
        async_b128(lB[0] + 64 * 32 * 2, gb + (size_t)64 * HH);                \
    }                                                                         \
    for (int it = 0; it < HH / 32; ++it) {                                    \
        int cur = it & 1;                                                     \
        if (it + 1 < HH / 32) {                                               \
            int nk = (it + 1) * 32, nb = (it + 1) & 1;                        \
            const bf16* ga = X + (size_t)(m0 + r0) * HH + nk + c0;            \
            const bf16* gb = W + (size_t)(n0 + r0) * HH + nk + c0;            \
            async_b128(lA[nb], ga);                                           \
            async_b128(lA[nb] + 64 * 32 * 2, ga + (size_t)64 * HH);           \
            async_b128(lB[nb], gb);                                           \
            async_b128(lB[nb] + 64 * 32 * 2, gb + (size_t)64 * HH);           \
            asm volatile("s_wait_asynccnt 0x4" ::: "memory");                 \
        } else {                                                              \
            asm volatile("s_wait_asynccnt 0x0" ::: "memory");                 \
        }                                                                     \
        __syncthreads();                                                      \
        v16bf af[2], bfr[4];                                                  \
        af[0] = load_a_frag32(sA[cur], wm * 32 + 0, lane, 32);                \
        af[1] = load_a_frag32(sA[cur], wm * 32 + 16, lane, 32);               \
        bfr[0] = load_b_frag32(sB[cur], wn * 64 + 0, lane, 32);               \
        bfr[1] = load_b_frag32(sB[cur], wn * 64 + 16, lane, 32);              \
        bfr[2] = load_b_frag32(sB[cur], wn * 64 + 32, lane, 32);              \
        bfr[3] = load_b_frag32(sB[cur], wn * 64 + 48, lane, 32);              \
        _Pragma("unroll")                                                     \
        for (int i = 0; i < 2; ++i)                                           \
            _Pragma("unroll")                                                 \
            for (int j = 0; j < 4; ++j)                                       \
                acc[i][j] = wmma_bf16(af[i], bfr[j], acc[i][j]);              \
        __syncthreads();                                                      \
    }

// Fused QKV projection. blockIdx.y in [0,24): sel = y>>3 picks Q/K/V,
// (y&7)*128 = N offset. Q,K -> [b,h,s,d]; V -> [b,h,d,s] (transposed).
__global__ void __launch_bounds__(256) gemm_qkv(
    const bf16* __restrict__ X, const bf16* __restrict__ Wq,
    const bf16* __restrict__ Wk, const bf16* __restrict__ Wv,
    const float* __restrict__ bq, const float* __restrict__ bk,
    const float* __restrict__ bv,
    bf16* __restrict__ Q, bf16* __restrict__ K, bf16* __restrict__ Vt) {
    int sel = blockIdx.y >> 3;
    int n0  = (blockIdx.y & 7) * 128;
    int m0  = blockIdx.x * 128;
    const bf16*  W    = (sel == 0) ? Wq : (sel == 1) ? Wk : Wv;
    const float* bias = (sel == 0) ? bq : (sel == 1) ? bk : bv;

    GEMM_CORE(X, W, m0, n0, acc)

    int half = lane >> 4, nn = lane & 15;
#pragma unroll
    for (int i = 0; i < 2; ++i)
#pragma unroll
        for (int j = 0; j < 4; ++j)
#pragma unroll
            for (int r = 0; r < 8; ++r) {
                int m = m0 + wm * 32 + 16 * i + r + 8 * half;
                int n = n0 + wn * 64 + 16 * j + nn;
                float v = acc[i][j][r] + bias[n];
                int b = m >> 11, s = m & (SS - 1);
                int head = n >> 6, d = n & 63;
                size_t bh = (size_t)b * NHH + head;
                if (sel == 0)
                    Q[(bh * SS + s) * DHH + d] = (bf16)v;
                else if (sel == 1)
                    K[(bh * SS + s) * DHH + d] = (bf16)v;
                else
                    Vt[(bh * DHH + d) * SS + s] = (bf16)v;
            }
}

// Output projection: x = ctx @ Wo^T + bo + hidden  (fp32 scratch for LN).
__global__ void __launch_bounds__(256) gemm_out(
    const bf16* __restrict__ X, const bf16* __restrict__ W,
    const float* __restrict__ bias, const float* __restrict__ resid,
    float* __restrict__ out) {
    int n0 = blockIdx.y * 128;
    int m0 = blockIdx.x * 128;

    GEMM_CORE(X, W, m0, n0, acc)

    int half = lane >> 4, nn = lane & 15;
#pragma unroll
    for (int i = 0; i < 2; ++i)
#pragma unroll
        for (int j = 0; j < 4; ++j)
#pragma unroll
            for (int r = 0; r < 8; ++r) {
                int m = m0 + wm * 32 + 16 * i + r + 8 * half;
                int n = n0 + wn * 64 + 16 * j + nn;
                size_t idx = (size_t)m * HH + n;
                out[idx] = acc[i][j][r] + bias[n] + resid[idx];
            }
}

// ---------------------------------------------------------------------------
// Flash attention. One wave per (head-batch, 16-query tile).
// grid = (S/16, B*NH), block = 32.
// ---------------------------------------------------------------------------
__global__ void __launch_bounds__(32) attn_flash(
    const bf16* __restrict__ Q, const bf16* __restrict__ K,
    const bf16* __restrict__ Vt, const float* __restrict__ amask,
    bf16* __restrict__ ctx) {
    __shared__ alignas(32) bf16 sP[16 * 32];

    int lane = threadIdx.x & 31;
    int half = lane >> 4, ln = lane & 15;
    int qt = blockIdx.x;                 // query tile
    int bh = blockIdx.y;                 // b*NH + head
    int b = bh >> 4, head = bh & 15;

    const bf16* Qb = Q + (size_t)bh * SS * DHH;
    const bf16* Kb = K + (size_t)bh * SS * DHH;
    const bf16* Vb = Vt + (size_t)bh * DHH * SS;
    const float* mb = amask + (size_t)b * SS;

    // Q A-fragments (K-dim = d = 64 -> two 16x32 frags), loaded once.
    const bf16* qrow = Qb + (size_t)(qt * 16 + ln) * DHH;
    v8bf q00 = *(const v8bf*)(qrow + 16 * half);
    v8bf q01 = *(const v8bf*)(qrow + 8 + 16 * half);
    v8bf q10 = *(const v8bf*)(qrow + 32 + 16 * half);
    v8bf q11 = *(const v8bf*)(qrow + 40 + 16 * half);
    v16bf aq0 = __builtin_shufflevector(q00, q01, 0, 1, 2, 3, 4, 5, 6, 7,
                                        8, 9, 10, 11, 12, 13, 14, 15);
    v16bf aq1 = __builtin_shufflevector(q10, q11, 0, 1, 2, 3, 4, 5, 6, 7,
                                        8, 9, 10, 11, 12, 13, 14, 15);

    v8f ctxa[4] = {};
    float mrow[8], lrow[8];
#pragma unroll
    for (int r = 0; r < 8; ++r) { mrow[r] = -1e30f; lrow[r] = 0.0f; }

    for (int j0 = 0; j0 < SS; j0 += 32) {
        // K fragments: rows of K[b,h,s,d] contiguous in d (B^T layout).
        const bf16* k0row = Kb + (size_t)(j0 + ln) * DHH;
        const bf16* k1row = k0row + 16 * DHH;
        v16bf bk00 = *(const v16bf*)(k0row + 16 * half);
        v16bf bk01 = *(const v16bf*)(k0row + 32 + 16 * half);
        v16bf bk10 = *(const v16bf*)(k1row + 16 * half);
        v16bf bk11 = *(const v16bf*)(k1row + 32 + 16 * half);

        v8f s0 = {}, s1 = {};
        s0 = wmma_bf16(aq0, bk00, s0);
        s0 = wmma_bf16(aq1, bk01, s0);
        s1 = wmma_bf16(aq0, bk10, s1);
        s1 = wmma_bf16(aq1, bk11, s1);

        float mk0 = mb[j0 + ln];
        float mk1 = mb[j0 + 16 + ln];

        float alpha[8];
#pragma unroll
        for (int r = 0; r < 8; ++r) {
            float a = s0[r] * 0.125f + mk0;   // 1/sqrt(64)
            float c = s1[r] * 0.125f + mk1;
            float mx = fmaxf(a, c);           // row max over 16-lane half
            mx = fmaxf(mx, __shfl_xor(mx, 1));
            mx = fmaxf(mx, __shfl_xor(mx, 2));
            mx = fmaxf(mx, __shfl_xor(mx, 4));
            mx = fmaxf(mx, __shfl_xor(mx, 8));
            float mnew = fmaxf(mrow[r], mx);
            float p0 = __expf(a - mnew);
            float p1 = __expf(c - mnew);
            s0[r] = p0;
            s1[r] = p1;
            float sum = p0 + p1;
            sum += __shfl_xor(sum, 1);
            sum += __shfl_xor(sum, 2);
            sum += __shfl_xor(sum, 4);
            sum += __shfl_xor(sum, 8);
            float al = __expf(mrow[r] - mnew);
            alpha[r] = al;
            lrow[r] = lrow[r] * al + sum;
            mrow[r] = mnew;
        }
#pragma unroll
        for (int db = 0; db < 4; ++db)
#pragma unroll
            for (int r = 0; r < 8; ++r) ctxa[db][r] *= alpha[r];

        // P (C-layout) -> LDS -> A-layout bf16 fragment
        __syncthreads();
#pragma unroll
        for (int r = 0; r < 8; ++r) {
            sP[(r + 8 * half) * 32 + ln]      = (bf16)s0[r];
            sP[(r + 8 * half) * 32 + 16 + ln] = (bf16)s1[r];
        }
        __syncthreads();
        v16bf ap = load_a_frag32(sP, 0, lane, 32);

        // ctx += P . V : V^T rows [b,h,d,s] contiguous in s (B^T layout)
#pragma unroll
        for (int db = 0; db < 4; ++db) {
            const bf16* vrow = Vb + (size_t)(db * 16 + ln) * SS + j0 + 16 * half;
            v16bf bv = *(const v16bf*)vrow;
            ctxa[db] = wmma_bf16(ap, bv, ctxa[db]);
        }
    }

    // finalize: ctx / l, store bf16 to [b, s, head*64 + d]
#pragma unroll
    for (int db = 0; db < 4; ++db)
#pragma unroll
        for (int r = 0; r < 8; ++r) {
            float v = ctxa[db][r] / lrow[r];
            int s = qt * 16 + r + 8 * half;
            int col = head * DHH + db * 16 + ln;
            ctx[((size_t)b * SS + s) * HH + col] = (bf16)v;
        }
}

// ---------------------------------------------------------------------------
// LayerNorm over H=1024 per row (population variance, eps=1e-12)
// ---------------------------------------------------------------------------
__global__ void __launch_bounds__(256) layernorm_row(
    const float* __restrict__ x, const float* __restrict__ gamma,
    const float* __restrict__ beta, float* __restrict__ out) {
    __shared__ float ws1[8], ws2[8];
    int row = blockIdx.x;
    const float* xr = x + (size_t)row * HH;
    float s = 0.f, s2 = 0.f;
    for (int i = threadIdx.x; i < HH; i += 256) {
        float v = xr[i];
        s += v;
        s2 += v * v;
    }
#pragma unroll
    for (int m = 1; m < 32; m <<= 1) {
        s += __shfl_xor(s, m);
        s2 += __shfl_xor(s2, m);
    }
    int wid = threadIdx.x >> 5, lane = threadIdx.x & 31;
    if (lane == 0) { ws1[wid] = s; ws2[wid] = s2; }
    __syncthreads();
    s = 0.f; s2 = 0.f;
#pragma unroll
    for (int w = 0; w < 8; ++w) { s += ws1[w]; s2 += ws2[w]; }
    float mu = s * (1.0f / HH);
    float var = s2 * (1.0f / HH) - mu * mu;
    float rstd = rsqrtf(var + 1e-12f);
    for (int i = threadIdx.x; i < HH; i += 256) {
        float v = (xr[i] - mu) * rstd;
        out[(size_t)row * HH + i] = v * gamma[i] + beta[i];
    }
}

// ---------------------------------------------------------------------------
extern "C" void kernel_launch(void* const* d_in, const int* in_sizes, int n_in,
                              void* d_out, int out_size, void* d_ws,
                              size_t ws_size, hipStream_t stream) {
    (void)in_sizes; (void)n_in; (void)out_size; (void)ws_size;
    const float* hidden = (const float*)d_in[0];
    const float* amask  = (const float*)d_in[1];
    const float* Wq     = (const float*)d_in[2];
    const float* bq     = (const float*)d_in[3];
    const float* Wk     = (const float*)d_in[4];
    const float* bk     = (const float*)d_in[5];
    const float* Wv     = (const float*)d_in[6];
    const float* bv     = (const float*)d_in[7];
    const float* Wo     = (const float*)d_in[8];
    const float* bo     = (const float*)d_in[9];
    const float* gamma  = (const float*)d_in[10];
    const float* beta   = (const float*)d_in[11];
    float* out = (float*)d_out;

    const size_t nMH = (size_t)MM * HH;     // 8388608
    const size_t nHH = (size_t)HH * HH;     // 1048576

    char* w = (char*)d_ws;
    bf16* hs_bf = (bf16*)w;  w += nMH * sizeof(bf16);
    bf16* wq_bf = (bf16*)w;  w += nHH * sizeof(bf16);
    bf16* wk_bf = (bf16*)w;  w += nHH * sizeof(bf16);
    bf16* wv_bf = (bf16*)w;  w += nHH * sizeof(bf16);
    bf16* wo_bf = (bf16*)w;  w += nHH * sizeof(bf16);
    bf16* q_ws  = (bf16*)w;  w += nMH * sizeof(bf16);
    bf16* k_ws  = (bf16*)w;  w += nMH * sizeof(bf16);
    bf16* vt_ws = (bf16*)w;  w += nMH * sizeof(bf16);
    bf16* ctx_ws = (bf16*)w; w += nMH * sizeof(bf16);
    float* x_ws = (float*)w; w += nMH * sizeof(float);

    cvt_f32_bf16<<<(int)(nMH / 256), 256, 0, stream>>>(hidden, hs_bf, (int)nMH);
    cvt_f32_bf16<<<(int)(nHH / 256), 256, 0, stream>>>(Wq, wq_bf, (int)nHH);
    cvt_f32_bf16<<<(int)(nHH / 256), 256, 0, stream>>>(Wk, wk_bf, (int)nHH);
    cvt_f32_bf16<<<(int)(nHH / 256), 256, 0, stream>>>(Wv, wv_bf, (int)nHH);
    cvt_f32_bf16<<<(int)(nHH / 256), 256, 0, stream>>>(Wo, wo_bf, (int)nHH);

    gemm_qkv<<<dim3(MM / 128, 24), 256, 0, stream>>>(
        hs_bf, wq_bf, wk_bf, wv_bf, bq, bk, bv, q_ws, k_ws, vt_ws);

    attn_flash<<<dim3(SS / 16, BB * NHH), 32, 0, stream>>>(
        q_ws, k_ws, vt_ws, amask, ctx_ws);

    gemm_out<<<dim3(MM / 128, 8), 256, 0, stream>>>(
        ctx_ws, wo_bf, bo, hidden, x_ws);

    layernorm_row<<<MM, 256, 0, stream>>>(x_ws, gamma, beta, out);
}
